// SelfAttention_37572373905865
// MI455X (gfx1250) — compile-verified, matrix operand order checked by reading, not run on hardware
//
#include <hip/hip_runtime.h>
#include <cstdint>

typedef __attribute__((ext_vector_type(16))) _Float16     v16h;
typedef __attribute__((ext_vector_type(8)))  float        v8f;
typedef __attribute__((ext_vector_type(4)))  unsigned int u32x4;
typedef __attribute__((ext_vector_type(8)))  int          i32x8;
typedef __attribute__((ext_vector_type(4)))  int          i32x4;

#define B2    128
#define NN    256
#define DD    256
#define HH    8
#define CC    32
#define HC    256   // H*C
#define QKV_E 768   // 3*H*C
#define OUTD  256

// ---------------------------------------------------------------------------
// WMMA 16x16x32 f16 fragment helpers (wave32, layouts per CDNA5 ISA 7.12.2)
// A (16xK row-major) / B (Kx16 column-major per lane): lane l covers row/col
// r = l&15; 16 halves map to k = kb + {0..7, 16..23}, kb = (l>>4)*8.
// C/D f32: value vi lives at row = vi + 8*(l>>4), col = l&15.
// ---------------------------------------------------------------------------
__device__ __forceinline__ v8f zero8() {
  v8f z;
#pragma unroll
  for (int j = 0; j < 8; ++j) z[j] = 0.0f;
  return z;
}

// element j -> src[r*stride + k]; per-lane halves are two contiguous 16B runs
__device__ __forceinline__ v16h frag_ld_f16(const _Float16* __restrict__ src,
                                            int stride, int lane) {
  const int r  = lane & 15;
  const int kb = (lane >> 4) * 8;
  const _Float16* p = src + (size_t)r * stride + kb;
  v16h f;
#pragma unroll
  for (int j = 0; j < 8; ++j) f[j] = p[j];
#pragma unroll
  for (int j = 0; j < 8; ++j) f[8 + j] = p[16 + j];
  return f;
}

__device__ __forceinline__ v8f wmma16(v16h a, v16h b, v8f c) {
  return __builtin_amdgcn_wmma_f32_16x16x32_f16(false, a, false, b,
                                                (short)0, c, false, false);
}

// ---------------------------------------------------------------------------
// TDM: 1-D contiguous f16 tile, global -> LDS.  D# per CDNA5 ISA ch.8:
//  g0: count=1 | lds_addr | global_addr[56:0] | type=2
//  g1: data_size=1 (2B), tensor_dim0=nelem, tensor_dim1=1, tile_dim0=nelem,
//      tile_dim1/2=0 (1-D), dim0_stride=nelem
//  g2/g3: zero (<=2-D tensor)
// Tracked by TENSORcnt; EXEC ignored -> issue from one wave only.
// ---------------------------------------------------------------------------
__device__ __forceinline__ void tdm_load_f16_1d(uint32_t lds_off,
                                                const _Float16* gsrc,
                                                uint32_t nelem) {
  const uint64_t ga = (uint64_t)(uintptr_t)gsrc;
  u32x4 g0;
  g0[0] = 1u;                                                  // count=1
  g0[1] = lds_off;                                             // lds_addr
  g0[2] = (uint32_t)ga;                                        // ga[31:0]
  g0[3] = (uint32_t)((ga >> 32) & 0x01FFFFFFu) | (2u << 30);   // ga[56:32]|type
  i32x8 g1;
  g1[0] = 0x00010000;                          // data_size=1 (2 bytes)
  g1[1] = (int)((nelem & 0xFFFFu) << 16);      // tensor_dim0[15:0]
  g1[2] = (int)(((nelem >> 16) & 0xFFFFu) | (1u << 16));  // dim0[31:16],dim1=1
  g1[3] = (int)((nelem & 0xFFFFu) << 16);      // tile_dim0
  g1[4] = 0;                                   // tile_dim1/2 = 0
  g1[5] = (int)nelem;                          // tensor_dim0_stride[31:0]
  g1[6] = 0;
  g1[7] = 0;
  i32x4 gz;
  gz[0] = gz[1] = gz[2] = gz[3] = 0;
#if defined(__clang_major__) && __clang_major__ >= 23
  i32x8 g4;
#pragma unroll
  for (int j = 0; j < 8; ++j) g4[j] = 0;
  __builtin_amdgcn_tensor_load_to_lds(g0, g1, gz, gz, g4, 0);
#else
  __builtin_amdgcn_tensor_load_to_lds(g0, g1, gz, gz, 0);
#endif
}

// ---------------------------------------------------------------------------
// Kernel 0: f32 -> f16 conversion pass (8 elems/thread, b128 in / b128 out)
// ---------------------------------------------------------------------------
__global__ __launch_bounds__(256) void k_cvt(const float* __restrict__ src,
                                             _Float16* __restrict__ dst, int n) {
  const int i = (blockIdx.x * 256 + threadIdx.x) * 8;
  if (i + 7 < n) {
#pragma unroll
    for (int j = 0; j < 8; ++j) dst[i + j] = (_Float16)src[i + j];
  }
}

// ---------------------------------------------------------------------------
// Kernel 1: fused QKV + gate projection, all-f16 operands.
//   x16[b2*n, d] x w16^T[1024, d]  (rows: 768 qkv, then 256 gate)
// Wave tile 32x64 (2 A-frags x 4 B-frags -> 8 WMMA); block = 2x4 waves
// = 64 rows x 256 cols.  Q pre-scaled, V stored transposed [b2,h,c,n],
// gate = sigmoid(.+bias) stored f16.
// ---------------------------------------------------------------------------
__global__ __launch_bounds__(256) void k_qkv_gate(
    const _Float16* __restrict__ x16,   // [B2*N, D]
    const _Float16* __restrict__ w16,   // [1024, D]
    const float* __restrict__ g_bias,   // [256]
    _Float16* __restrict__ q_ws,        // [B2,H,N,C] (scaled)
    _Float16* __restrict__ k_ws,        // [B2,H,N,C]
    _Float16* __restrict__ vT_ws,       // [B2,H,C,N]
    _Float16* __restrict__ gate_ws) {   // [B2,N,HC]
  const int lane = threadIdx.x & 31;
  const int wv   = threadIdx.x >> 5;
  const int wr   = wv >> 2;            // 0..1
  const int wc   = wv & 3;             // 0..3
  const int b2   = blockIdx.z;
  const int row0 = blockIdx.y * 64 + wr * 32;
  const int col0 = blockIdx.x * 256 + wc * 64;
  const int rl   = (lane >> 4) * 8;
  const int cl   = lane & 15;

  const _Float16* xrow = x16 + ((size_t)b2 * NN + row0) * DD;
  __builtin_prefetch(xrow, 0, 3);

  v8f acc[2][4];
#pragma unroll
  for (int m = 0; m < 2; ++m)
#pragma unroll
    for (int t = 0; t < 4; ++t) acc[m][t] = zero8();

  for (int kk = 0; kk < DD; kk += 32) {
    v16h a0 = frag_ld_f16(xrow + kk, DD, lane);
    v16h a1 = frag_ld_f16(xrow + (size_t)16 * DD + kk, DD, lane);
#pragma unroll
    for (int t = 0; t < 4; ++t) {
      v16h b = frag_ld_f16(w16 + (size_t)(col0 + t * 16) * DD + kk, DD, lane);
      acc[0][t] = wmma16(a0, b, acc[0][t]);
      acc[1][t] = wmma16(a1, b, acc[1][t]);
    }
  }

#pragma unroll
  for (int mt = 0; mt < 2; ++mt) {
    const int rowb = row0 + mt * 16;
#pragma unroll
    for (int t = 0; t < 4; ++t) {
      const int nb = col0 + t * 16;   // tile-uniform fused column
      const int n  = nb + cl;
      if (nb < QKV_E) {
        const int which = nb / HC;    // 0=q 1=k 2=v (block-uniform here)
        const int hc = n - which * HC;
        const int h = hc >> 5, c = hc & 31;
        if (which == 2) {             // V transposed: [b2,h,c,n]
          const size_t base = (((size_t)b2 * HH + h) * CC + c) * NN;
#pragma unroll
          for (int vi = 0; vi < 8; ++vi)
            vT_ws[base + rowb + vi + rl] = (_Float16)acc[mt][t][vi];
        } else {
          _Float16* dst = which ? k_ws : q_ws;
          const float scale = which ? 1.0f : 0.17677669529663687f;  // C^-1/2
          const size_t base = (((size_t)b2 * HH + h) * NN) * CC + c;
#pragma unroll
          for (int vi = 0; vi < 8; ++vi)
            dst[base + (size_t)(rowb + vi + rl) * CC] =
                (_Float16)(acc[mt][t][vi] * scale);
        }
      } else {
        const int g = n - QKV_E;
        const float gb = g_bias[g];
        const size_t base = ((size_t)b2 * NN) * HC + g;
#pragma unroll
        for (int vi = 0; vi < 8; ++vi) {
          const float z = acc[mt][t][vi] + gb;
          gate_ws[base + (size_t)(rowb + vi + rl) * HC] =
              (_Float16)(1.0f / (1.0f + __expf(-z)));
        }
      }
    }
  }
}

// ---------------------------------------------------------------------------
// Kernel 2: attention per (b2, h).  TDM stages K (16 KB) and V^T (16 KB)
// into LDS; 8 waves x 32 query rows; logits via WMMA; softmax via __shfl_xor
// (a row spans one 16-lane half); P staged to LDS in two 128-key halves
// (32 KB total -> LDS = 64 KB); PV via WMMA; sigmoid gate folded into store.
// ---------------------------------------------------------------------------
__global__ __launch_bounds__(256) void k_attn(
    const _Float16* __restrict__ q_ws,
    const _Float16* __restrict__ k_ws,
    const _Float16* __restrict__ vT_ws,
    const float* __restrict__ mask,       // [B2,N]
    const float* __restrict__ nb_bias,    // [H,N,N]
    const _Float16* __restrict__ gate_ws, // [B2,N,HC]
    _Float16* __restrict__ wa_ws) {       // [B2,N,HC] gated
  __shared__ _Float16 Klds[NN * CC];      // 16 KB  [key n][c]
  __shared__ _Float16 VTlds[CC * NN];     // 16 KB  [c][key n]
  __shared__ _Float16 Pb[8][16 * 128];    // 32 KB  per-wave P half-tile

  const int lane = threadIdx.x & 31;
  const int wv   = threadIdx.x >> 5;
  const int h    = blockIdx.x;
  const int b2   = blockIdx.y;
  const int rl   = (lane >> 4) * 8;
  const int cl   = lane & 15;

  const size_t hd = (((size_t)b2 * HH + h) * NN) * CC;

  if (wv == 0) {  // one wave issues both TDM descriptors
    tdm_load_f16_1d((uint32_t)(uintptr_t)&Klds[0],  k_ws + hd,  NN * CC);
    tdm_load_f16_1d((uint32_t)(uintptr_t)&VTlds[0], vT_ws + hd, NN * CC);
    __builtin_amdgcn_s_wait_tensorcnt(0);
  }
  __syncthreads();

  for (int mt = 0; mt < 2; ++mt) {
    const int m0 = wv * 32 + mt * 16;
    const v16h qa = frag_ld_f16(q_ws + hd + (size_t)m0 * CC, CC, lane);

    v8f lg[16];
#pragma unroll
    for (int nt = 0; nt < 16; ++nt) {
      v16h kf = frag_ld_f16(&Klds[nt * 16 * CC], CC, lane);  // K rows = B cols
      lg[nt] = wmma16(qa, kf, zero8());
    }

    // + mask bias (per key col) + nonbatched bias (per row,col)
#pragma unroll
    for (int nt = 0; nt < 16; ++nt) {
      const int n = nt * 16 + cl;
      const float mb = 1e9f * (mask[(size_t)b2 * NN + n] - 1.0f);
      const float* nbp = nb_bias + ((size_t)h * NN) * NN + n;
#pragma unroll
      for (int vi = 0; vi < 8; ++vi)
        lg[nt][vi] += mb + nbp[(size_t)(m0 + vi + rl) * NN];
    }

    // softmax: row (m0+vi+rl) lives across one 16-lane half at value vi
#pragma unroll
    for (int vi = 0; vi < 8; ++vi) {
      float mx = -3.0e38f;
#pragma unroll
      for (int nt = 0; nt < 16; ++nt) mx = fmaxf(mx, lg[nt][vi]);
      for (int off = 1; off < 16; off <<= 1) mx = fmaxf(mx, __shfl_xor(mx, off, 32));
      float s = 0.0f;
#pragma unroll
      for (int nt = 0; nt < 16; ++nt) {
        const float e = __expf(lg[nt][vi] - mx);
        lg[nt][vi] = e;
        s += e;
      }
      for (int off = 1; off < 16; off <<= 1) s += __shfl_xor(s, off, 32);
      const float inv = 1.0f / s;
#pragma unroll
      for (int nt = 0; nt < 16; ++nt) lg[nt][vi] *= inv;
    }

    // P (16x256) x V (256x32), keys processed in two 128-wide halves.
    // Same-wave LDS producer/consumer in lockstep -> no barrier needed.
    v8f ao0 = zero8(), ao1 = zero8();
    for (int half = 0; half < 2; ++half) {
#pragma unroll
      for (int nt = 0; nt < 8; ++nt) {
        const int ntg = half * 8 + nt;
#pragma unroll
        for (int vi = 0; vi < 8; ++vi)
          Pb[wv][(vi + rl) * 128 + nt * 16 + cl] = (_Float16)lg[ntg][vi];
      }
#pragma unroll
      for (int ks = 0; ks < 4; ++ks) {
        const int kbase = half * 128 + ks * 32;
        v16h pa = frag_ld_f16(&Pb[wv][ks * 32], 128, lane);
        v16h b0 = frag_ld_f16(&VTlds[kbase], NN, lane);            // c 0..15
        v16h b1 = frag_ld_f16(&VTlds[16 * NN + kbase], NN, lane);  // c 16..31
        ao0 = wmma16(pa, b0, ao0);
        ao1 = wmma16(pa, b1, ao1);
      }
    }

    // gate * wa -> f16 workspace
#pragma unroll
    for (int ct = 0; ct < 2; ++ct) {
      const v8f& ao = ct ? ao1 : ao0;
#pragma unroll
      for (int vi = 0; vi < 8; ++vi) {
        const int r = m0 + vi + rl;
        const int c = ct * 16 + cl;
        const size_t idx = ((size_t)b2 * NN + r) * HC + h * CC + c;
        wa_ws[idx] = (_Float16)(ao[vi] * (float)gate_ws[idx]);
      }
    }
  }
}

// ---------------------------------------------------------------------------
// Kernel 3: output projection  out[M,256] = wa[M,256] x wo16^T + b_o
// Wave tile 32x64; block = 2x4 waves = 64 rows x all 256 cols.
// ---------------------------------------------------------------------------
__global__ __launch_bounds__(256) void k_out(
    const _Float16* __restrict__ wa_ws,  // [B2*N, HC]
    const _Float16* __restrict__ wo16,   // [256, HC]
    const float* __restrict__ b_o,       // [256]
    float* __restrict__ out) {           // [B2*N, 256]
  const int lane = threadIdx.x & 31;
  const int wv   = threadIdx.x >> 5;
  const int wr   = wv >> 2;
  const int wc   = wv & 3;
  const int row0 = blockIdx.x * 64 + wr * 32;
  const int col0 = wc * 64;
  const int rl   = (lane >> 4) * 8;
  const int cl   = lane & 15;

  const _Float16* arow = wa_ws + (size_t)row0 * HC;
  __builtin_prefetch(arow, 0, 3);

  v8f acc[2][4];
#pragma unroll
  for (int m = 0; m < 2; ++m)
#pragma unroll
    for (int t = 0; t < 4; ++t) acc[m][t] = zero8();

  for (int kk = 0; kk < HC; kk += 32) {
    v16h a0 = frag_ld_f16(arow + kk, HC, lane);
    v16h a1 = frag_ld_f16(arow + (size_t)16 * HC + kk, HC, lane);
#pragma unroll
    for (int t = 0; t < 4; ++t) {
      v16h b = frag_ld_f16(wo16 + (size_t)(col0 + t * 16) * HC + kk, HC, lane);
      acc[0][t] = wmma16(a0, b, acc[0][t]);
      acc[1][t] = wmma16(a1, b, acc[1][t]);
    }
  }

#pragma unroll
  for (int mt = 0; mt < 2; ++mt) {
#pragma unroll
    for (int t = 0; t < 4; ++t) {
      const int o = col0 + t * 16 + cl;
      const float bo = b_o[o];
#pragma unroll
      for (int vi = 0; vi < 8; ++vi) {
        const int r = row0 + mt * 16 + vi + rl;
        out[(size_t)r * OUTD + o] = acc[mt][t][vi] + bo;
      }
    }
  }
}

// ---------------------------------------------------------------------------
extern "C" void kernel_launch(void* const* d_in, const int* in_sizes, int n_in,
                              void* d_out, int out_size, void* d_ws, size_t ws_size,
                              hipStream_t stream) {
  (void)in_sizes; (void)n_in; (void)out_size; (void)ws_size;
  const float* in_data = (const float*)d_in[0];
  const float* mask    = (const float*)d_in[1];
  const float* nb_bias = (const float*)d_in[2];
  const float* w_qkv   = (const float*)d_in[3];
  const float* w_gate  = (const float*)d_in[4];
  const float* g_bias  = (const float*)d_in[5];
  const float* w_o     = (const float*)d_in[6];
  const float* b_o     = (const float*)d_in[7];
  float* out = (float*)d_out;

  // Workspace layout (f16 elements)
  const size_t XN  = (size_t)B2 * NN * DD;       // 8,388,608
  const size_t QN  = (size_t)B2 * HH * NN * CC;  // 8,388,608
  _Float16* x16    = (_Float16*)d_ws;
  _Float16* w16    = x16 + XN;                   // [1024 x 256] fused qkv|gate
  _Float16* wo16   = w16 + (size_t)(QKV_E + HC) * DD;
  _Float16* q_ws   = wo16 + (size_t)OUTD * HC;
  _Float16* k_ws   = q_ws + QN;
  _Float16* vT_ws  = k_ws + QN;
  _Float16* gate16 = vT_ws + QN;
  _Float16* wa_ws  = gate16 + (size_t)B2 * NN * HC;

  // 0) one-time f32 -> f16 conversion passes
  k_cvt<<<dim3((int)(XN / 8 / 256)), 256, 0, stream>>>(in_data, x16, (int)XN);
  k_cvt<<<dim3(QKV_E * DD / 8 / 256), 256, 0, stream>>>(w_qkv, w16, QKV_E * DD);
  k_cvt<<<dim3(HC * DD / 8 / 256), 256, 0, stream>>>(w_gate, w16 + (size_t)QKV_E * DD, HC * DD);
  k_cvt<<<dim3(OUTD * HC / 8 / 256), 256, 0, stream>>>(w_o, wo16, OUTD * HC);

  // 1) fused QKV + gate projection: [32768 x 256] x [256 x 1024]
  k_qkv_gate<<<dim3(4, 4, B2), 256, 0, stream>>>(x16, w16, g_bias,
                                                 q_ws, k_ws, vT_ws, gate16);
  // 2) attention per (h, b2), K/V^T staged via Tensor Data Mover
  k_attn<<<dim3(HH, B2), 256, 0, stream>>>(q_ws, k_ws, vT_ws, mask, nb_bias,
                                           gate16, wa_ws);
  // 3) output projection: [32768 x 256] x [256 x 256] + bias
  k_out<<<dim3((B2 * NN) / 64), 256, 0, stream>>>(wa_ws, wo16, b_o, out);
}
